// CQAttn_25451976196156
// MI455X (gfx1250) — matrix operand order repaired and checked by reading, not compiled, not run on hardware
//
#include <hip/hip_runtime.h>
#include <hip/hip_bf16.h>

// ---------------------------------------------------------------------------
// CQ attention for MI455X (gfx1250, wave32, WMMA bf16 16x16x32, f32 accum)
// B=32, CL=2048, QL=128, H=768.  Memory-bound (~1.4 GB traffic @ 23.3 TB/s).
// Double-buffered LDS staging (1 barrier / k-step) + global_prefetch_b8 on
// the streaming c reads.
// ---------------------------------------------------------------------------

#define B_   32
#define CL_  2048
#define QL_  128
#define H_   768
#define NEG_INF_ -1.0e30f

typedef __attribute__((ext_vector_type(16))) __bf16 v16bf;
typedef __attribute__((ext_vector_type(8)))  float  v8f;

__device__ __forceinline__ v16bf ld_bf16x16(const __bf16* p0, const __bf16* p1) {
  // Two 16-byte chunks (8 halves each) -> one WMMA operand register set.
  union { uint4 u[2]; v16bf v; } t;
  t.u[0] = *(const uint4*)p0;
  t.u[1] = *(const uint4*)p1;
  return t.v;
}

__device__ __forceinline__ v8f wmma_bf16(v16bf a, v16bf b, v8f c) {
  return __builtin_amdgcn_wmma_f32_16x16x32_bf16(false, a, false, b, (short)0, c,
                                                 false, false);
}

// Order-preserving float->uint encoding so atomicMax(uint) == float max.
__device__ __forceinline__ unsigned int enc_f32(float f) {
  unsigned int u = __float_as_uint(f);
  return (u & 0x80000000u) ? ~u : (u | 0x80000000u);
}
__device__ __forceinline__ float dec_f32(unsigned int u) {
  unsigned int v = (u & 0x80000000u) ? (u & 0x7FFFFFFFu) : ~u;
  return __uint_as_float(v);
}

// ---------------------------------------------------------------------------
// k_init: zero the encoded column-max buffer (encoded 0 == most-negative).
// ---------------------------------------------------------------------------
__global__ void cq_init_colmax(unsigned int* colmax_u) {
  int i = blockIdx.x * blockDim.x + threadIdx.x;
  if (i < B_ * QL_) colmax_u[i] = 0u;
}

// ---------------------------------------------------------------------------
// k0: per (b, j) q row: qw = q * W_cq (bf16, row-major), qT = q^T (bf16),
//     S_qv = dot(q, W_q).
// ---------------------------------------------------------------------------
__global__ __launch_bounds__(256)
void cq_prep_q(const float* __restrict__ q, const float* __restrict__ W_cq,
               const float* __restrict__ W_q, __bf16* __restrict__ qw,
               __bf16* __restrict__ qT, float* __restrict__ S_qv) {
  const int bj = blockIdx.x;            // 0 .. B*QL-1
  const int b  = bj >> 7;
  const int j  = bj & (QL_ - 1);
  const int t  = threadIdx.x;
  const float* qrow = q + (size_t)bj * H_;

  float part = 0.f;
#pragma unroll
  for (int i = 0; i < 3; ++i) {
    int h = t + 256 * i;
    float v = qrow[h];
    qw[(size_t)bj * H_ + h] = (__bf16)(v * W_cq[h]);
    qT[((size_t)b * H_ + h) * QL_ + j] = (__bf16)v;
    part += v * W_q[h];
  }
  __shared__ float red[256];
  red[t] = part;
  __syncthreads();
  for (int s = 128; s > 0; s >>= 1) {
    if (t < s) red[t] += red[t + s];
    __syncthreads();
  }
  if (t == 0) S_qv[bj] = red[0];
}

// ---------------------------------------------------------------------------
// k1: logits S[b, c0:c0+32, 0:128] via WMMA over H (A = c bf16 staged in LDS,
//     B = qw rows), fold in S_c, S_q, bias; row softmax -> S1 (bf16);
//     raw S -> ws (f32); per-column block max -> atomicMax(colmax).
//     LDS staging is double-buffered: one barrier per 32-deep k-step.
// ---------------------------------------------------------------------------
__global__ __launch_bounds__(256)
void cq_logits(const float* __restrict__ c, const __bf16* __restrict__ qw,
               const float* __restrict__ W_c, const float* __restrict__ S_qv,
               const float* __restrict__ mask_c, const float* __restrict__ mask_q,
               const float* __restrict__ bias, float* __restrict__ S_raw,
               __bf16* __restrict__ S1, unsigned int* __restrict__ colmax_u) {
  const int b    = blockIdx.y;
  const int c0   = blockIdx.x * 32;
  const int t    = threadIdx.x;
  const int w    = t >> 5;          // wave 0..7
  const int lane = t & 31;
  const int p    = w & 1;           // m-half (rows 16p .. 16p+15)
  const int ncol = w >> 1;          // column group: cols 32*ncol .. +31

  __shared__ __bf16 c_lds[2][32][32];   // ping-pong staging (4 KB)
  __shared__ float  S_full[32][QL_];
  __shared__ float  sc_red[32][8];
  __shared__ float  row_red[32][8];
  __shared__ float  sqv_lds[QL_];
  __shared__ float  mq_lds[QL_];
  __shared__ float  mc_lds[32];

  if (t < QL_) {
    sqv_lds[t] = S_qv[b * QL_ + t] + bias[0];
    mq_lds[t]  = mask_q[b * QL_ + t];
  }
  if (t < 32) mc_lds[t] = mask_c[(size_t)b * CL_ + c0 + t];

  // staging assignment: 8 threads per row, 4 floats each per k-step
  const int srow = t >> 3;
  const int skc  = (t & 7) * 4;
  const float* crow_ptr = c + (size_t)(b * CL_ + c0 + srow) * H_;
  float sc_part = 0.f;

  // WMMA operand lane patterns (contiguous k = 16h + s, same for A and B)
  const int arow = p * 16 + (lane & 15);
  const int akb  = (lane < 16) ? 0 : 16;  // A: K run {akb..akb+15}
  const int bn0  = ncol * 32 + (lane & 15);
  const int bkb  = (lane < 16) ? 0 : 16;  // B: K run {bkb..bkb+15}

  v8f acc0 = {0.f, 0.f, 0.f, 0.f, 0.f, 0.f, 0.f, 0.f};
  v8f acc1 = {0.f, 0.f, 0.f, 0.f, 0.f, 0.f, 0.f, 0.f};

  auto stage = [&](int ks, int buf) {
    const int h0 = ks * 32;
    float4 cv = *(const float4*)(crow_ptr + h0 + skc);
    float4 wv = *(const float4*)(W_c + h0 + skc);
    sc_part += cv.x * wv.x + cv.y * wv.y + cv.z * wv.z + cv.w * wv.w;
    union { __bf16 h[4]; uint2 u; } pk;
    pk.h[0] = (__bf16)cv.x; pk.h[1] = (__bf16)cv.y;
    pk.h[2] = (__bf16)cv.z; pk.h[3] = (__bf16)cv.w;
    *(uint2*)&c_lds[buf][srow][skc] = pk.u;
  };

  stage(0, 0);
  __syncthreads();

  for (int ks = 0; ks < H_ / 32; ++ks) {
    const int cur = ks & 1;
    if (ks + 1 < H_ / 32) {                       // uniform branch (SCC only)
      __builtin_prefetch(crow_ptr + (ks + 2) * 32 + skc, 0, 1);
      stage(ks + 1, cur ^ 1);
    }
    const int h0 = ks * 32;
    v16bf a = ld_bf16x16(&c_lds[cur][arow][akb], &c_lds[cur][arow][akb + 8]);
    const __bf16* qwp = qw + ((size_t)(b * QL_ + bn0)) * H_ + h0 + bkb;
    v16bf b0 = ld_bf16x16(qwp, qwp + 8);
    v16bf b1 = ld_bf16x16(qwp + 16 * H_, qwp + 16 * H_ + 8);  // n+16 tile
    acc0 = wmma_bf16(a, b0, acc0);
    acc1 = wmma_bf16(a, b1, acc1);
    __syncthreads();   // publishes buf cur^1, protects buf cur for next stage
  }

  // reduce S_c (8 partials per row)
  sc_red[srow][t & 7] = sc_part;
  __syncthreads();
  if (t < 32) {
    float s = 0.f;
#pragma unroll
    for (int i = 0; i < 8; ++i) s += sc_red[t][i];
    sc_red[t][0] = s;
  }
  __syncthreads();

  // compose full logits tile in LDS: S = S_cq + S_c + S_qv (+bias)
  {
    const int rbase = p * 16 + ((lane < 16) ? 0 : 8);
    const int cA = ncol * 32 + (lane & 15);
    const int cB = cA + 16;
#pragma unroll
    for (int r = 0; r < 8; ++r) {
      int row = rbase + r;
      float scv = sc_red[row][0];
      S_full[row][cA] = acc0[r] + scv + sqv_lds[cA];
      S_full[row][cB] = acc1[r] + scv + sqv_lds[cB];
    }
  }
  __syncthreads();

  // row softmax (mask_q), write S1 bf16 and raw S f32
  {
    const int row = t >> 3;
    const int cb  = (t & 7) * 16;
    float lmax = NEG_INF_;
#pragma unroll
    for (int i = 0; i < 16; ++i) {
      float v = (mq_lds[cb + i] > 0.f) ? S_full[row][cb + i] : NEG_INF_;
      lmax = fmaxf(lmax, v);
    }
    row_red[row][t & 7] = lmax;
    __syncthreads();
    float rmax = row_red[row][0];
#pragma unroll
    for (int i = 1; i < 8; ++i) rmax = fmaxf(rmax, row_red[row][i]);
    __syncthreads();

    float evals[16];
    float lsum = 0.f;
#pragma unroll
    for (int i = 0; i < 16; ++i) {
      float v = (mq_lds[cb + i] > 0.f) ? S_full[row][cb + i] : NEG_INF_;
      float e = __expf(v - rmax);
      evals[i] = e;
      lsum += e;
    }
    row_red[row][t & 7] = lsum;
    __syncthreads();
    float rsum = 0.f;
#pragma unroll
    for (int i = 0; i < 8; ++i) rsum += row_red[row][i];
    float inv = 1.f / rsum;

    const size_t orow = (size_t)(b * CL_ + c0 + row) * QL_ + cb;
    float4* srp = (float4*)(S_raw + orow);
#pragma unroll
    for (int i = 0; i < 4; ++i) {
      float4 v;
      v.x = S_full[row][cb + 4 * i + 0];
      v.y = S_full[row][cb + 4 * i + 1];
      v.z = S_full[row][cb + 4 * i + 2];
      v.w = S_full[row][cb + 4 * i + 3];
      srp[i] = v;
    }
#pragma unroll
    for (int i = 0; i < 16; ++i) S1[orow + i] = (__bf16)(evals[i] * inv);
  }
  __syncthreads();

  // per-column max over this 32-row tile (mask_c) -> atomic float-max
  if (t < QL_) {
    float m = NEG_INF_;
#pragma unroll 4
    for (int r2 = 0; r2 < 32; ++r2)
      if (mc_lds[r2] > 0.f) m = fmaxf(m, S_full[r2][t]);
    atomicMax(&colmax_u[b * QL_ + t], enc_f32(m));
  }
}

// ---------------------------------------------------------------------------
// k2: Q2^T[b, h, j] = (1/colsum_j) * sum_k exp(S[b,k,j]-colmax_j) * c[b,k,h]
//     One 16(j) x 128(h) output block per CTA, K loop over CL=2048.
//     Double-buffered LDS staging: one barrier per 32-deep k-step.
// ---------------------------------------------------------------------------
__global__ __launch_bounds__(256)
void cq_q2(const float* __restrict__ c, const float* __restrict__ S_raw,
           const unsigned int* __restrict__ colmax_u,
           const float* __restrict__ mask_c, __bf16* __restrict__ Q2T) {
  const int b    = blockIdx.z;
  const int j0   = blockIdx.y * 16;
  const int h0   = blockIdx.x * 128;
  const int t    = threadIdx.x;
  const int w    = t >> 5;
  const int lane = t & 31;

  __shared__ __bf16 cT[2][128][32];    // [buf][h local][k local]  (16 KB)
  __shared__ __bf16 e_lds[2][16][32];  // [buf][j local][k local]  (2 KB)
  __shared__ float  cm[16];
  __shared__ float  cs_red[16][16];
  __shared__ float  cs[16];

  if (t < 16) cm[t] = dec_f32(colmax_u[b * QL_ + j0 + t]);
  __syncthreads();

  const int ej  = t & 15;            // exp-staging j (fixed per thread)
  const int ckk = t >> 3;            // c-staging k (0..31)
  const int chb = (t & 7) * 16;      // c-staging h base

  const int aj  = lane & 15;
  const int akb = (lane < 16) ? 0 : 16;
  const int bn  = 16 * w + (lane & 15);
  const int bkb = (lane < 16) ? 0 : 16;

  float csum = 0.f;
  v8f acc = {0.f, 0.f, 0.f, 0.f, 0.f, 0.f, 0.f, 0.f};

  auto stage = [&](int k0, int buf) {
    // exp tile (16 j x 32 k), mask_c applied; accumulate colsum
#pragma unroll
    for (int rep = 0; rep < 2; ++rep) {
      int kk = (t >> 4) + rep * 16;
      float s = S_raw[((size_t)(b * CL_) + k0 + kk) * QL_ + j0 + ej];
      float m = mask_c[(size_t)b * CL_ + k0 + kk];
      float e = (m > 0.f) ? __expf(s - cm[ej]) : 0.f;
      e_lds[buf][ej][kk] = (__bf16)e;
      csum += e;
    }
    // c^T tile (128 h x 32 k)
    const float* cp = c + ((size_t)(b * CL_) + k0 + ckk) * H_ + h0 + chb;
#pragma unroll
    for (int i = 0; i < 4; ++i) {
      float4 v = ((const float4*)cp)[i];
      cT[buf][chb + 4 * i + 0][ckk] = (__bf16)v.x;
      cT[buf][chb + 4 * i + 1][ckk] = (__bf16)v.y;
      cT[buf][chb + 4 * i + 2][ckk] = (__bf16)v.z;
      cT[buf][chb + 4 * i + 3][ckk] = (__bf16)v.w;
    }
  };

  stage(0, 0);
  __syncthreads();

  for (int ks = 0; ks < CL_ / 32; ++ks) {
    const int cur = ks & 1;
    if (ks + 1 < CL_ / 32) {                      // uniform branch
      __builtin_prefetch(
          c + ((size_t)(b * CL_) + (ks + 2) * 32 + ckk) * H_ + h0 + chb, 0, 1);
      stage((ks + 1) * 32, cur ^ 1);
    }
    v16bf a  = ld_bf16x16(&e_lds[cur][aj][akb], &e_lds[cur][aj][akb + 8]);
    v16bf bm = ld_bf16x16(&cT[cur][bn][bkb], &cT[cur][bn][bkb + 8]);
    acc = wmma_bf16(a, bm, acc);
    __syncthreads();
  }

  // reduce column sums (16 partials per j)
  cs_red[ej][t >> 4] = csum;
  __syncthreads();
  if (t < 16) {
    float s = 0.f;
#pragma unroll
    for (int i = 0; i < 16; ++i) s += cs_red[t][i];
    cs[t] = s;
  }
  __syncthreads();

  const int jr = (lane < 16) ? 0 : 8;
#pragma unroll
  for (int r = 0; r < 8; ++r) {
    int j = jr + r;
    float v = acc[r] / cs[j];
    Q2T[((size_t)b * H_ + h0 + 16 * w + (lane & 15)) * QL_ + j0 + j] = (__bf16)v;
  }
}

// ---------------------------------------------------------------------------
// k3: A = S1 @ q, Bmat = S1 @ Q2 (K=QL=128), fused output
//     G = [c, A, c*A, c*Bmat]  (B, CL, 4H) f32.
// ---------------------------------------------------------------------------
__global__ __launch_bounds__(256)
void cq_out(const float* __restrict__ c, const __bf16* __restrict__ S1,
            const __bf16* __restrict__ qT, const __bf16* __restrict__ Q2T,
            float* __restrict__ G) {
  const int b    = blockIdx.y;
  const int c0   = blockIdx.x * 32;
  const int t    = threadIdx.x;
  const int w    = t >> 5;
  const int lane = t & 31;
  const int p    = w & 1;

  __shared__ __bf16 s1_lds[32][QL_];  // 8 KB
  {
    const uint4* src = (const uint4*)(S1 + (size_t)(b * CL_ + c0) * QL_);
    uint4* dst = (uint4*)&s1_lds[0][0];
    dst[t]       = src[t];
    dst[t + 256] = src[t + 256];
  }
  __syncthreads();

  const int arow  = p * 16 + (lane & 15);
  const int akb   = (lane < 16) ? 0 : 16;
  const int n     = lane & 15;
  const int bkb   = (lane < 16) ? 0 : 16;
  const int rbase = p * 16 + ((lane < 16) ? 0 : 8);

  for (int i = 0; i < 12; ++i) {
    const int h0 = 16 * ((w >> 1) + 4 * i);
    // hint the c row segment needed at epilogue into cache
    __builtin_prefetch(c + ((size_t)(b * CL_) + c0 + rbase) * H_ + h0 + n, 0, 1);
    v8f accA = {0.f, 0.f, 0.f, 0.f, 0.f, 0.f, 0.f, 0.f};
    v8f accB = {0.f, 0.f, 0.f, 0.f, 0.f, 0.f, 0.f, 0.f};
    const __bf16* qTp = qT  + ((size_t)b * H_ + h0 + n) * QL_;
    const __bf16* q2p = Q2T + ((size_t)b * H_ + h0 + n) * QL_;
#pragma unroll
    for (int ks = 0; ks < 4; ++ks) {
      const int kb32 = ks * 32;
      v16bf a  = ld_bf16x16(&s1_lds[arow][kb32 + akb],
                            &s1_lds[arow][kb32 + akb + 8]);
      v16bf b1 = ld_bf16x16(qTp + kb32 + bkb, qTp + kb32 + bkb + 8);
      v16bf b2 = ld_bf16x16(q2p + kb32 + bkb, q2p + kb32 + bkb + 8);
      accA = wmma_bf16(a, b1, accA);
      accB = wmma_bf16(a, b2, accB);
    }
#pragma unroll
    for (int r = 0; r < 8; ++r) {
      const int crow = c0 + rbase + r;
      const int hcol = h0 + n;
      float cv = c[((size_t)(b * CL_) + crow) * H_ + hcol];
      float av = accA[r];
      float bv = accB[r];
      float* gp = G + ((size_t)(b * CL_) + crow) * (4 * H_) + hcol;
      gp[0]        = cv;
      gp[H_]       = av;
      gp[2 * H_]   = cv * av;
      gp[3 * H_]   = cv * bv;
    }
  }
}

// ---------------------------------------------------------------------------
// host launcher
// ---------------------------------------------------------------------------
extern "C" void kernel_launch(void* const* d_in, const int* in_sizes, int n_in,
                              void* d_out, int out_size, void* d_ws, size_t ws_size,
                              hipStream_t stream) {
  const float* c      = (const float*)d_in[0];
  const float* q      = (const float*)d_in[1];
  const float* mask_c = (const float*)d_in[2];
  const float* mask_q = (const float*)d_in[3];
  const float* W_c    = (const float*)d_in[4];
  const float* W_q    = (const float*)d_in[5];
  const float* W_cq   = (const float*)d_in[6];
  const float* bias   = (const float*)d_in[7];
  float* G = (float*)d_out;

  char* ws = (char*)d_ws;
  float*        S_raw  = (float*)ws;        ws += (size_t)B_ * CL_ * QL_ * 4;
  __bf16*       S1     = (__bf16*)ws;       ws += (size_t)B_ * CL_ * QL_ * 2;
  __bf16*       qw     = (__bf16*)ws;       ws += (size_t)B_ * QL_ * H_ * 2;
  __bf16*       qT     = (__bf16*)ws;       ws += (size_t)B_ * H_ * QL_ * 2;
  __bf16*       Q2T    = (__bf16*)ws;       ws += (size_t)B_ * H_ * QL_ * 2;
  float*        S_qv   = (float*)ws;        ws += (size_t)B_ * QL_ * 4;
  unsigned int* colmax = (unsigned int*)ws; ws += (size_t)B_ * QL_ * 4;

  cq_init_colmax<<<dim3(16), dim3(256), 0, stream>>>(colmax);
  cq_prep_q<<<dim3(B_ * QL_), dim3(256), 0, stream>>>(q, W_cq, W_q, qw, qT, S_qv);
  cq_logits<<<dim3(CL_ / 32, B_), dim3(256), 0, stream>>>(
      c, qw, W_c, S_qv, mask_c, mask_q, bias, S_raw, S1, colmax);
  cq_q2<<<dim3(H_ / 128, QL_ / 16, B_), dim3(256), 0, stream>>>(
      c, S_raw, colmax, mask_c, Q2T);
  cq_out<<<dim3(CL_ / 32, B_), dim3(256), 0, stream>>>(c, S1, qT, Q2T, G);
}
